// StateSpaceModel_12738873000150
// MI455X (gfx1250) — matrix operand order repaired and checked by reading, not compiled, not run on hardware
//
#include <hip/hip_runtime.h>
#include <hip/hip_bf16.h>
#include <math.h>

// ---------------- types for WMMA ----------------
typedef __bf16 bf16_t;
typedef __attribute__((ext_vector_type(16))) __bf16 v16bf;
typedef __attribute__((ext_vector_type(8)))  float  v8f;

union FragU {
  struct { uint4 q0, q1; } q;
  v16bf v;
};

__device__ __forceinline__ v16bf load_frag16(const bf16_t* p0, const bf16_t* p1) {
  FragU u;
  u.q.q0 = *(const uint4*)p0;
  u.q.q1 = *(const uint4*)p1;
  return u.v;
}

__device__ __forceinline__ v8f wmma_bf16(v16bf a, v16bf b, v8f c) {
  // (neg_a, A, neg_b, B, c_mod, C, reuse_a, reuse_b)
  return __builtin_amdgcn_wmma_f32_16x16x32_bf16(false, a, false, b, (short)0, c,
                                                 false, false);
}

__device__ __forceinline__ float gelu_f(float x) {
  // tanh approximation (jax.nn.gelu default)
  float c = 0.7978845608028654f * (x + 0.044715f * x * x * x);
  return 0.5f * x * (1.0f + tanhf(c));
}
__device__ __forceinline__ float silu_f(float x) {
  return x / (1.0f + __expf(-x));
}

// ---------------- conversion kernels ----------------
__global__ void f32_to_bf16_kernel(const float* __restrict__ src,
                                   bf16_t* __restrict__ dst, size_t n) {
  size_t i = (size_t)blockIdx.x * blockDim.x + threadIdx.x;
  if (i < n) dst[i] = (bf16_t)src[i];
}

// conv weight (Cout, Cin, k) f32 -> Wt[(t*Cin+ci)][co] bf16 (K x N row-major)
__global__ void convw_transpose_kernel(const float* __restrict__ w,
                                       bf16_t* __restrict__ wt,
                                       int Cout, int Cin, int kw) {
  size_t i = (size_t)blockIdx.x * blockDim.x + threadIdx.x;
  size_t tot = (size_t)Cout * Cin * kw;
  if (i >= tot) return;
  int t = i % kw;
  int ci = (i / kw) % Cin;
  int co = i / ((size_t)kw * Cin);
  wt[((size_t)t * Cin + ci) * Cout + co] = (bf16_t)w[i];
}

__global__ void copy_f32_kernel(const float* __restrict__ s, float* __restrict__ d,
                                size_t n) {
  size_t i = (size_t)blockIdx.x * blockDim.x + threadIdx.x;
  if (i < n) d[i] = s[i];
}

// ---------------- conv layer 0 (Cin=1, k=10, s=5) + GELU -> bf16 ----------------
__global__ void conv0_gelu_kernel(const float* __restrict__ src,
                                  const float* __restrict__ w,
                                  const float* __restrict__ bias,
                                  bf16_t* __restrict__ out, int Lout, int T) {
  size_t tid = (size_t)blockIdx.x * blockDim.x + threadIdx.x;
  size_t tot = (size_t)8 * Lout * 512;
  if (tid >= tot) return;
  int c = tid & 511;
  size_t r = tid >> 9;
  int l = r % Lout;
  int b = r / Lout;
  const float* sp = src + (size_t)b * T + (size_t)l * 5;
  float s = bias[c];
#pragma unroll
  for (int t = 0; t < 10; ++t) s += w[c * 10 + t] * sp[t];
  out[tid] = (bf16_t)gelu_f(s);
}

// ---------------- conv layers 1..6 as implicit-im2col WMMA GEMM ----------------
// Each wave computes a 16 x (16*NT) output tile: A fragment is loaded once per
// k-step and reused against NT B fragments (4 wmma per A-load when NT=4).
// act layout (B, L, 512) bf16.  Row m = b*Lout + l; im2col row = act + (b*Lin + l*s)*512.
template <int NT>
__global__ void convgemm_wmma_kernel(const bf16_t* __restrict__ actIn, int Lin,
                                     int Lout, int stride,
                                     const bf16_t* __restrict__ Wt,
                                     const float* __restrict__ bias,
                                     bf16_t* __restrict__ actOut,
                                     float* __restrict__ outF, int Ktot) {
  int lane = threadIdx.x & 31;
  int wid = threadIdx.x >> 5;
  int M = 8 * Lout;
  int tiles_m = (M + 15) >> 4;
  const int groups_n = 512 / (16 * NT);
  int tile = blockIdx.x * (blockDim.x >> 5) + wid;
  if (tile >= tiles_m * groups_n) return;
  int tm = tile / groups_n, tg = tile % groups_n;
  int n0 = tg * 16 * NT;

  int mrow = tm * 16 + (lane & 15);
  if (mrow > M - 1) mrow = M - 1;
  int b = mrow / Lout, l = mrow % Lout;
  const bf16_t* arow =
      actIn + ((size_t)b * Lin + (size_t)l * stride) * 512 + 8 * (lane >> 4);
  const bf16_t* brow = Wt + (size_t)(8 * (lane >> 4)) * 512 + n0 + (lane & 15);

  v8f acc[NT];
#pragma unroll
  for (int j = 0; j < NT; ++j) acc[j] = (v8f){};

  for (int k0 = 0; k0 < Ktot; k0 += 32) {
    v16bf af = load_frag16(arow + k0, arow + k0 + 16);
    __builtin_prefetch((const void*)(arow + k0 + 256), 0, 1);
    const bf16_t* bp = brow + (size_t)k0 * 512;
#pragma unroll
    for (int j = 0; j < NT; ++j) {
      v16bf bf;
#pragma unroll
      for (int s = 0; s < 16; ++s) bf[s] = bp[(size_t)(s < 8 ? s : s + 8) * 512 + 16 * j];
      acc[j] = wmma_bf16(af, bf, acc[j]);
    }
  }

  int mbase = tm * 16 + ((lane >> 4) << 3);
#pragma unroll
  for (int j = 0; j < NT; ++j) {
    int n = n0 + 16 * j + (lane & 15);
    float bv = bias[n];
#pragma unroll
    for (int v = 0; v < 8; ++v) {
      int mr = mbase + v;
      if (mr < M) {
        float g = gelu_f(acc[j][v] + bv);
        actOut[(size_t)mr * 512 + n] = (bf16_t)g;
        if (outF) outF[(size_t)mr * 512 + n] = g;
      }
    }
  }
}

// ---------------- generic WMMA GEMM:  C = act(A @ B + bias) ----------------
// A: (M,K) bf16 lda;  B: (K,N) bf16 ldb;  act: 0 none, 1 softplus
// Wave computes 16 x (16*NT) tile; requires N % (16*NT) == 0.
template <int NT>
__global__ void gemm_wmma_kernel(const bf16_t* __restrict__ A, int lda,
                                 const bf16_t* __restrict__ Bw, int ldb,
                                 const float* __restrict__ bias, float* Cf,
                                 bf16_t* Cb, int ldc, int M, int N, int K, int act,
                                 int accum) {
  int lane = threadIdx.x & 31;
  int wid = threadIdx.x >> 5;
  int groups_n = N / (16 * NT);
  int tiles_m = (M + 15) >> 4;
  int tile = blockIdx.x * (blockDim.x >> 5) + wid;
  if (tile >= tiles_m * groups_n) return;
  int tm = tile / groups_n, tg = tile % groups_n;
  int n0 = tg * 16 * NT;

  int mrow = tm * 16 + (lane & 15);
  if (mrow > M - 1) mrow = M - 1;
  const bf16_t* arow = A + (size_t)mrow * lda + 8 * (lane >> 4);
  const bf16_t* brow = Bw + (size_t)(8 * (lane >> 4)) * ldb + n0 + (lane & 15);

  v8f acc[NT];
#pragma unroll
  for (int j = 0; j < NT; ++j) acc[j] = (v8f){};

  for (int k0 = 0; k0 < K; k0 += 32) {
    v16bf af = load_frag16(arow + k0, arow + k0 + 16);
    __builtin_prefetch((const void*)(arow + k0 + 256), 0, 1);
    const bf16_t* bp = brow + (size_t)k0 * ldb;
#pragma unroll
    for (int j = 0; j < NT; ++j) {
      v16bf bf;
#pragma unroll
      for (int s = 0; s < 16; ++s) bf[s] = bp[(size_t)(s < 8 ? s : s + 8) * ldb + 16 * j];
      acc[j] = wmma_bf16(af, bf, acc[j]);
    }
  }

  int mbase = tm * 16 + ((lane >> 4) << 3);
#pragma unroll
  for (int j = 0; j < NT; ++j) {
    int n = n0 + 16 * j + (lane & 15);
    float bv = bias ? bias[n] : 0.0f;
#pragma unroll
    for (int v = 0; v < 8; ++v) {
      int mr = mbase + v;
      if (mr < M) {
        float val = acc[j][v] + bv;
        if (act == 1) val = (val > 20.0f) ? val : log1pf(__expf(val));
        size_t idx = (size_t)mr * ldc + n;
        if (Cf) {
          if (accum) Cf[idx] += val; else Cf[idx] = val;
        }
        if (Cb) Cb[idx] = (bf16_t)val;
      }
    }
  }
}

// ---------------- layernorm: one wave per row of 512 ----------------
__global__ void layernorm_kernel(const float* __restrict__ x,
                                 const float* __restrict__ w,
                                 const float* __restrict__ bb, float* outF,
                                 bf16_t* outB, int M, int N) {
  int lane = threadIdx.x & 31;
  int wid = threadIdx.x >> 5;
  int row = blockIdx.x * (blockDim.x >> 5) + wid;
  if (row >= M) return;
  const float* xr = x + (size_t)row * N;
  float s = 0.f, s2 = 0.f;
  for (int c = lane; c < N; c += 32) {
    float v = xr[c];
    s += v;
    s2 += v * v;
  }
#pragma unroll
  for (int o = 16; o > 0; o >>= 1) {
    s += __shfl_xor(s, o);
    s2 += __shfl_xor(s2, o);
  }
  float mu = s / N;
  float var = s2 / N - mu * mu;
  float inv = rsqrtf(var + 1e-5f);
  for (int c = lane; c < N; c += 32) {
    float v = (xr[c] - mu) * inv * w[c] + bb[c];
    if (outF) outF[(size_t)row * N + c] = v;
    if (outB) outB[(size_t)row * N + c] = (bf16_t)v;
  }
}

// ---------------- depthwise causal/anticausal conv (k=4) + SiLU ----------------
// xz (rows, 2048) f32; xc channels are cols [0,1024).  dir=0 fwd, dir=1 bwd.
__global__ void dwconv_silu_kernel(const float* __restrict__ xz,
                                   const float* __restrict__ w,
                                   const float* __restrict__ cb,
                                   float* __restrict__ xc_f,
                                   bf16_t* __restrict__ xc_b, int L, int dir) {
  size_t tid = (size_t)blockIdx.x * blockDim.x + threadIdx.x;
  size_t tot = (size_t)8 * L * 1024;
  if (tid >= tot) return;
  int d = tid & 1023;
  size_t r = tid >> 10;
  int u = r % L;
  int b = r / L;
  float s = cb[d];
#pragma unroll
  for (int j = 0; j < 4; ++j) {
    int uu = dir ? (u + 3 - j) : (u - 3 + j);
    if (uu >= 0 && uu < L)
      s += w[d * 4 + j] * xz[((size_t)b * L + uu) * 2048 + d];
  }
  float v = silu_f(s);
  xc_f[tid] = v;
  xc_b[tid] = (bf16_t)v;
}

// ---------------- selective scan (8 states), per (b, channel) thread ----------
__global__ void scan_kernel(const float* __restrict__ dt,
                            const float* __restrict__ xdbl,
                            const float* __restrict__ xc_f,
                            const float* __restrict__ xz,
                            const float* __restrict__ A_log,
                            const float* __restrict__ Dskip,
                            bf16_t* __restrict__ y_b, int L, int dir) {
  int tid = blockIdx.x * blockDim.x + threadIdx.x;
  if (tid >= 8 * 1024) return;
  int b = tid >> 10;
  int d = tid & 1023;
  float Arow[8];
#pragma unroll
  for (int n = 0; n < 8; ++n) Arow[n] = -__expf(A_log[d * 8 + n]);
  float Dv = Dskip[d];
  float h[8] = {0.f, 0.f, 0.f, 0.f, 0.f, 0.f, 0.f, 0.f};
  for (int t = 0; t < L; ++t) {
    int u = dir ? (L - 1 - t) : t;
    size_t r = (size_t)b * L + u;
    float dtv = dt[r * 1024 + d];
    float xcv = xc_f[r * 1024 + d];
    float zv = xz[r * 2048 + 1024 + d];
    const float* bc = xdbl + r * 48;
    float dx = dtv * xcv;
    float acc = 0.f;
#pragma unroll
    for (int n = 0; n < 8; ++n) {
      float a = __expf(dtv * Arow[n]);
      h[n] = a * h[n] + dx * bc[32 + n];
      acc += h[n] * bc[40 + n];
    }
    float y = (acc + xcv * Dv) * silu_f(zv);
    y_b[r * 1024 + d] = (bf16_t)y;
  }
}

// ---------------- decoder: sigmoid(x @ w + b), one wave per row ----------------
__global__ void decoder_kernel(const float* __restrict__ x,
                               const float* __restrict__ w,
                               const float* __restrict__ b, float* __restrict__ out,
                               int M, int N) {
  int lane = threadIdx.x & 31;
  int wid = threadIdx.x >> 5;
  int row = blockIdx.x * (blockDim.x >> 5) + wid;
  if (row >= M) return;
  float s = 0.f;
  for (int c = lane; c < N; c += 32) s += x[(size_t)row * N + c] * w[c];
#pragma unroll
  for (int o = 16; o > 0; o >>= 1) s += __shfl_xor(s, o);
  if (lane == 0) out[row] = 1.0f / (1.0f + __expf(-(s + b[0])));
}

// ==================== host side ====================
static inline void launch_gemm(hipStream_t s, const bf16_t* A, int lda,
                               const bf16_t* B, int ldb, const float* bias,
                               float* Cf, bf16_t* Cb, int ldc, int M, int N, int K,
                               int act, int accum) {
  int tiles_m = (M + 15) / 16;
  if (N % 64 == 0) {
    int tiles = tiles_m * (N / 64);
    gemm_wmma_kernel<4><<<(tiles + 3) / 4, 128, 0, s>>>(A, lda, B, ldb, bias, Cf,
                                                        Cb, ldc, M, N, K, act,
                                                        accum);
  } else {  // N == 48
    int tiles = tiles_m * (N / 48);
    gemm_wmma_kernel<3><<<(tiles + 3) / 4, 128, 0, s>>>(A, lda, B, ldb, bias, Cf,
                                                        Cb, ldc, M, N, K, act,
                                                        accum);
  }
}

extern "C" void kernel_launch(void* const* d_in, const int* in_sizes, int n_in,
                              void* d_out, int out_size, void* d_ws, size_t ws_size,
                              hipStream_t stream) {
  (void)in_sizes; (void)n_in; (void)out_size; (void)ws_size;
  const int Ls[8] = {80000, 15999, 7999, 3999, 1999, 999, 499, 249};
  const int ks[7] = {10, 3, 3, 3, 3, 2, 2};
  const int ss[7] = {5, 2, 2, 2, 2, 2, 2};
  const int Lf = 249, Mr = 8 * Lf;  // 1992 rows in mamba stage

  // ---- input pointers (setup_inputs dict order, flattened) ----
  const float* src = (const float*)d_in[0];
  const float *cw[7], *cb[7];
  for (int i = 0; i < 7; ++i) {
    cw[i] = (const float*)d_in[1 + 2 * i];
    cb[i] = (const float*)d_in[2 + 2 * i];
  }
  struct MB {
    const float *in_proj, *conv_w, *conv_b, *x_proj, *dt_w, *dt_b, *A_log, *Dskip,
        *out_proj;
  };
  const float *ln_w[2], *ln_b[2], *pln_w[2], *pln_b[2];
  MB mb[4];  // layer0-fwd, layer0-bwd, layer1-fwd, layer1-bwd
  for (int j = 0; j < 2; ++j) {
    int base = 15 + 22 * j;
    ln_w[j] = (const float*)d_in[base + 0];
    ln_b[j] = (const float*)d_in[base + 1];
    for (int dir = 0; dir < 2; ++dir) {
      int o = base + 2 + 9 * dir;
      MB& m = mb[j * 2 + dir];
      m.in_proj = (const float*)d_in[o + 0];
      m.conv_w = (const float*)d_in[o + 1];
      m.conv_b = (const float*)d_in[o + 2];
      m.x_proj = (const float*)d_in[o + 3];
      m.dt_w = (const float*)d_in[o + 4];
      m.dt_b = (const float*)d_in[o + 5];
      m.A_log = (const float*)d_in[o + 6];
      m.Dskip = (const float*)d_in[o + 7];
      m.out_proj = (const float*)d_in[o + 8];
    }
    pln_w[j] = (const float*)d_in[base + 20];
    pln_b[j] = (const float*)d_in[base + 21];
  }
  const float* dec_w = (const float*)d_in[59];
  const float* dec_b = (const float*)d_in[60];

  // ---- workspace arena ----
  char* base = (char*)d_ws;
  size_t off = 0;
  auto alloc = [&](size_t bytes) -> char* {
    off = (off + 255) & ~(size_t)255;
    char* p = base + off;
    off += bytes;
    return p;
  };
  bf16_t* actA = (bf16_t*)alloc(2ull * 8 * Ls[1] * 512);
  bf16_t* actB = (bf16_t*)alloc(2ull * 8 * Ls[2] * 512);
  bf16_t* wt[7];
  for (int i = 1; i <= 6; ++i) wt[i] = (bf16_t*)alloc(2ull * ks[i] * 512 * 512);
  struct MW { bf16_t *inp, *xp, *dtw, *outp; } mw[4];
  for (int k = 0; k < 4; ++k) {
    mw[k].inp = (bf16_t*)alloc(2ull * 512 * 2048);
    mw[k].xp = (bf16_t*)alloc(2ull * 1024 * 48);
    mw[k].dtw = (bf16_t*)alloc(2ull * 32 * 1024);
    mw[k].outp = (bf16_t*)alloc(2ull * 1024 * 512);
  }
  float* x_f = (float*)alloc(4ull * Mr * 512);
  float* x_acc = (float*)alloc(4ull * Mr * 512);
  bf16_t* xn_b = (bf16_t*)alloc(2ull * Mr * 512);
  float* xz = (float*)alloc(4ull * Mr * 2048);
  float* xc_f = (float*)alloc(4ull * Mr * 1024);
  bf16_t* xc_b = (bf16_t*)alloc(2ull * Mr * 1024);
  float* xdbl_f = (float*)alloc(4ull * Mr * 48);
  bf16_t* xdbl_b = (bf16_t*)alloc(2ull * Mr * 48);
  float* dt_f = (float*)alloc(4ull * Mr * 1024);
  bf16_t* y_b = (bf16_t*)alloc(2ull * Mr * 1024);

  // ---- weight conversions (every call; deterministic) ----
  for (int i = 1; i <= 6; ++i) {
    size_t n = (size_t)512 * 512 * ks[i];
    convw_transpose_kernel<<<(n + 255) / 256, 256, 0, stream>>>(cw[i], wt[i], 512,
                                                                512, ks[i]);
  }
  for (int k = 0; k < 4; ++k) {
    size_t n;
    n = 512ull * 2048;
    f32_to_bf16_kernel<<<(n + 255) / 256, 256, 0, stream>>>(mb[k].in_proj,
                                                            mw[k].inp, n);
    n = 1024ull * 48;
    f32_to_bf16_kernel<<<(n + 255) / 256, 256, 0, stream>>>(mb[k].x_proj, mw[k].xp,
                                                            n);
    n = 32ull * 1024;
    f32_to_bf16_kernel<<<(n + 255) / 256, 256, 0, stream>>>(mb[k].dt_w, mw[k].dtw,
                                                            n);
    n = 1024ull * 512;
    f32_to_bf16_kernel<<<(n + 255) / 256, 256, 0, stream>>>(mb[k].out_proj,
                                                            mw[k].outp, n);
  }

  // ---- conv frontend ----
  {
    size_t tot = (size_t)8 * Ls[1] * 512;
    conv0_gelu_kernel<<<(tot + 255) / 256, 256, 0, stream>>>(src, cw[0], cb[0],
                                                             actA, Ls[1], Ls[0]);
  }
  bf16_t* bufs[2] = {actA, actB};
  int cur = 0;
  for (int i = 1; i <= 6; ++i) {
    bf16_t* in = bufs[cur];
    bf16_t* out = bufs[cur ^ 1];
    int M = 8 * Ls[i + 1];
    int tiles = ((M + 15) / 16) * 8;  // groups_n = 512/64
    float* outF = (i == 6) ? x_f : nullptr;
    convgemm_wmma_kernel<4><<<(tiles + 3) / 4, 128, 0, stream>>>(
        in, Ls[i], Ls[i + 1], ss[i], wt[i], cb[i], out, outF, ks[i] * 512);
    cur ^= 1;
  }

  // ---- bidirectional mamba layers ----
  for (int j = 0; j < 2; ++j) {
    layernorm_kernel<<<(Mr + 3) / 4, 128, 0, stream>>>(x_f, ln_w[j], ln_b[j],
                                                       nullptr, xn_b, Mr, 512);
    {
      size_t n = (size_t)Mr * 512;
      copy_f32_kernel<<<(n + 255) / 256, 256, 0, stream>>>(x_f, x_acc, n);
    }
    for (int dir = 0; dir < 2; ++dir) {
      int k = j * 2 + dir;
      // xz = xn @ in_proj : (1992,512)x(512,2048)
      launch_gemm(stream, xn_b, 512, mw[k].inp, 2048, nullptr, xz, nullptr, 2048,
                  Mr, 2048, 512, 0, 0);
      // depthwise conv (causal / anticausal) + silu
      {
        size_t tot = (size_t)Mr * 1024;
        dwconv_silu_kernel<<<(tot + 255) / 256, 256, 0, stream>>>(
            xz, mb[k].conv_w, mb[k].conv_b, xc_f, xc_b, Lf, dir);
      }
      // x_dbl = xc @ x_proj : (1992,1024)x(1024,48), keep f32 + bf16
      launch_gemm(stream, xc_b, 1024, mw[k].xp, 48, nullptr, xdbl_f, xdbl_b, 48,
                  Mr, 48, 1024, 0, 0);
      // dt = softplus(x_dbl[:, :32] @ dt_w + dt_b) : (1992,32)x(32,1024)
      launch_gemm(stream, xdbl_b, 48, mw[k].dtw, 1024, mb[k].dt_b, dt_f, nullptr,
                  1024, Mr, 1024, 32, 1, 0);
      // selective scan + gate -> y (bf16)
      scan_kernel<<<32, 256, 0, stream>>>(dt_f, xdbl_f, xc_f, xz, mb[k].A_log,
                                          mb[k].Dskip, y_b, Lf, dir);
      // x_acc += y @ out_proj : (1992,1024)x(1024,512)
      launch_gemm(stream, y_b, 1024, mw[k].outp, 512, nullptr, x_acc, nullptr,
                  512, Mr, 512, 1024, 0, 1);
    }
    layernorm_kernel<<<(Mr + 3) / 4, 128, 0, stream>>>(x_acc, pln_w[j], pln_b[j],
                                                       x_f, nullptr, Mr, 512);
  }

  // ---- decoder ----
  decoder_kernel<<<(Mr + 3) / 4, 128, 0, stream>>>(x_f, dec_w, dec_b,
                                                   (float*)d_out, Mr, 512);
}
